// GCN_73091753443469
// MI455X (gfx1250) — compile-verified, matrix operand order checked by reading, not run on hardware
//
#include <hip/hip_runtime.h>
#include <hip/hip_bf16.h>

typedef __attribute__((ext_vector_type(16))) _Float16 v16h;
typedef __attribute__((ext_vector_type(8)))  float    v8f;

#define IN_D  128
#define HID   64

// ---------------- degree / norm ----------------

__global__ void deg_init_kernel(float* deg, int n) {
    int i = blockIdx.x * blockDim.x + threadIdx.x;
    if (i < n) deg[i] = 1.0f;               // self loop contributes 1 to deg[dst]
}

__global__ void deg_accum_kernel(const int* __restrict__ dst, float* deg, int e) {
    int i = blockIdx.x * blockDim.x + threadIdx.x;
    if (i < e) unsafeAtomicAdd(&deg[dst[i]], 1.0f);
}

__global__ void dinv_kernel(float* deg, int n) {
    int i = blockIdx.x * blockDim.x + threadIdx.x;
    if (i < n) deg[i] = rsqrtf(deg[i]);     // deg >= 1 always (self loop)
}

// ---------------- weight packing into WMMA B-fragment layout ----------------
// Fragment f = ks*4 + ct (ks = k0/32 step, ct = 16-col tile). Per lane: 16 f16 (32 B).
// B fragment layout mirrors the 16-bit A layout with N in place of M:
// lane l -> col n = l&15, koff = (l>>4)*8; halves j=0..7 -> K=k0+koff+j, 8..15 -> K=k0+16+koff+j.

template <int K>
__global__ void pack_w_kernel(const float* __restrict__ W, _Float16* __restrict__ Wp) {
    const int frag = (blockIdx.x * blockDim.x + threadIdx.x) >> 5;
    const int lane = threadIdx.x & 31;
    const int nFrag = (K / 32) * 4;
    if (frag >= nFrag) return;
    const int ks = frag >> 2;
    const int ct = frag & 3;
    const int n    = lane & 15;
    const int koff = (lane >> 4) << 3;
    v16h b;
    const float* wp = W + (size_t)(ks * 32 + koff) * HID + ct * 16 + n;
#pragma unroll
    for (int j = 0; j < 8; ++j) {
        b[j]     = (_Float16)wp[(size_t)j * HID];
        b[8 + j] = (_Float16)wp[(size_t)(16 + j) * HID];
    }
    *(v16h*)(Wp + ((size_t)frag * 32 + lane) * 16) = b;
}

// ---------------- dense GEMM: hW = (relu?)(A) @ W ----------------
// One wave32 computes a 16-row x 64-col strip: 4 accumulators, A fragment reused
// across the 4 column tiles; B fragments fetched pre-packed (2x b128 per fragment).

template <int K, bool RELU>
__global__ void gemm_wmma_kernel(const float* __restrict__ A,
                                 const _Float16* __restrict__ Wp,
                                 float* __restrict__ out, int nRows) {
    const int wave = (blockIdx.x * blockDim.x + threadIdx.x) >> 5;
    const int lane = threadIdx.x & 31;
    const int rowTiles = nRows >> 4;        // nRows = 100000 = 6250*16
    if (wave >= rowTiles) return;

    const int m    = lane & 15;
    const int koff = (lane >> 4) << 3;      // 0 or 8
    const int row  = (wave << 4) + m;

    const v16h* bp = (const v16h*)Wp;       // [nFrag][32] fragments

    v8f acc0 = {}, acc1 = {}, acc2 = {}, acc3 = {};

#pragma unroll
    for (int k0 = 0; k0 < K; k0 += 32) {
        const int ks = k0 >> 5;
        // A fragment: 8 + 8 consecutive f32 -> f16
        v16h a;
        const float* ap = A + (size_t)row * K + k0 + koff;
#pragma unroll
        for (int j = 0; j < 8; ++j) {
            float f0 = ap[j];
            float f1 = ap[16 + j];
            if (RELU) { f0 = fmaxf(f0, 0.0f); f1 = fmaxf(f1, 0.0f); }
            a[j]     = (_Float16)f0;
            a[8 + j] = (_Float16)f1;
        }
#pragma unroll
        for (int ct = 0; ct < 4; ++ct) {
            v16h b = bp[(size_t)(ks * 4 + ct) * 32 + lane];
            v8f& acc = (ct == 0 ? acc0 : ct == 1 ? acc1 : ct == 2 ? acc2 : acc3);
            acc = __builtin_amdgcn_wmma_f32_16x16x32_f16(
                false, a, false, b, (short)0, acc, false, false);
        }
    }

    // C/D layout: vgpr i -> lanes0-15: M=i, N=lane; lanes16-31: M=8+i, N=lane-16
    const int colLane = lane & 15;
    const int rBase   = (wave << 4) + ((lane >> 4) << 3);
    auto store_tile = [&](v8f acc, int ct) {
#pragma unroll
        for (int i = 0; i < 8; ++i)
            out[(size_t)(rBase + i) * HID + ct * 16 + colLane] = acc[i];
    };
    store_tile(acc0, 0); store_tile(acc1, 1); store_tile(acc2, 2); store_tile(acc3, 3);
}

// ---------------- aggregation ----------------
// out[n][c] = hW[n][c] * dinv[n]^2 (self loop) + bias[c]
__global__ void agg_init_kernel(const float* __restrict__ hW,
                                const float* __restrict__ dinv,
                                const float* __restrict__ bias,
                                float* __restrict__ out, int total) {
    int idx = blockIdx.x * blockDim.x + threadIdx.x;
    if (idx >= total) return;
    int node = idx >> 6;
    int c    = idx & 63;
    float w = dinv[node];
    float v = hW[idx] * (w * w);
    if (bias) v += bias[c];
    out[idx] = v;
}

// out[dst] += hW[src] * dinv[src]*dinv[dst]; one wave per edge, 2 floats per lane
__global__ void agg_edges_kernel(const float* __restrict__ hW,
                                 const int* __restrict__ src,
                                 const int* __restrict__ dst,
                                 const float* __restrict__ dinv,
                                 float* __restrict__ out, int nEdges) {
    const int wave = (blockIdx.x * blockDim.x + threadIdx.x) >> 5;
    const int lane = threadIdx.x & 31;
    if (wave >= nEdges) return;
    const int s = src[wave];
    const int d = dst[wave];
    const float w = dinv[s] * dinv[d];
    const float2 v = ((const float2*)(hW + (size_t)s * HID))[lane];
    float* op = out + (size_t)d * HID + lane * 2;
    unsafeAtomicAdd(op,     v.x * w);
    unsafeAtomicAdd(op + 1, v.y * w);
}

// ---------------- host-side orchestration ----------------

extern "C" void kernel_launch(void* const* d_in, const int* in_sizes, int n_in,
                              void* d_out, int out_size, void* d_ws, size_t ws_size,
                              hipStream_t stream) {
    const float* x     = (const float*)d_in[0];   // [N, 128]
    const int*   eidx  = (const int*)d_in[1];     // [2, E]
    const float* W_in  = (const float*)d_in[2];   // [128, 64]
    const float* b_in  = (const float*)d_in[3];   // [64]
    const float* W_h   = (const float*)d_in[4];   // [3, 64, 64]
    const float* b_h   = (const float*)d_in[5];   // [3, 64]
    const float* W_out = (const float*)d_in[6];   // [64, 64]

    const int N = in_sizes[0] / IN_D;             // 100000
    const int E = in_sizes[1] / 2;                // 1600000
    const int L = in_sizes[4] / (HID * HID);      // 3

    const int* src = eidx;
    const int* dst = eidx + E;

    // Workspace layout (256B-aligned chunks):
    //   dinv [N] f32 | B1 [N*64] f32 | packed weights: W_in(16KB), W_h[0..2](8KB ea), W_out(8KB)
    char* ws = (char*)d_ws;
    auto align256 = [](size_t v) { return (v + 255) & ~(size_t)255; };
    size_t off = 0;
    float* dinv = (float*)(ws + off);  off = align256(off + (size_t)N * sizeof(float));
    float* B1   = (float*)(ws + off);  off = align256(off + (size_t)N * HID * sizeof(float));
    _Float16* WpIn  = (_Float16*)(ws + off);  off = align256(off + 16 * 1024);
    _Float16* WpH0  = (_Float16*)(ws + off);  off = align256(off + 8 * 1024);
    _Float16* WpH1  = (_Float16*)(ws + off);  off = align256(off + 8 * 1024);
    _Float16* WpH2  = (_Float16*)(ws + off);  off = align256(off + 8 * 1024);
    _Float16* WpOut = (_Float16*)(ws + off);  off = align256(off + 8 * 1024);
    _Float16* WpHs[3] = {WpH0, WpH1, WpH2};
    float* O = (float*)d_out;                     // ping buffer; final result lands here

    const int T = 256;

    // 1) degrees -> dinv (in place)
    deg_init_kernel<<<(N + T - 1) / T, T, 0, stream>>>(dinv, N);
    deg_accum_kernel<<<(E + T - 1) / T, T, 0, stream>>>(dst, dinv, E);
    dinv_kernel<<<(N + T - 1) / T, T, 0, stream>>>(dinv, N);

    // 2) pack all weight matrices into WMMA fragment layout (f16)
    pack_w_kernel<IN_D><<<2, T, 0, stream>>>(W_in, WpIn);            // 16 fragments
    for (int l = 0; l < L; ++l)
        pack_w_kernel<HID><<<1, T, 0, stream>>>(W_h + (size_t)l * HID * HID, WpHs[l]);
    pack_w_kernel<HID><<<1, T, 0, stream>>>(W_out, WpOut);           // 8 fragments

    const int rowTiles   = N / 16;                            // 6250
    const int gemmBlocks = (rowTiles + 7) / 8;                // 8 waves / block
    const int totalNC    = N * HID;
    const int initBlocks = (totalNC + T - 1) / T;
    const int edgeBlocks = (E + 7) / 8;                       // 8 waves / block

    // conv1: x @ W_in, +b_in, no relu on input
    gemm_wmma_kernel<IN_D, false><<<gemmBlocks, T, 0, stream>>>(x, WpIn, B1, N);
    agg_init_kernel<<<initBlocks, T, 0, stream>>>(B1, dinv, b_in, O, totalNC);
    agg_edges_kernel<<<edgeBlocks, T, 0, stream>>>(B1, src, dst, dinv, O, E);

    // hidden convs: layer0 input = conv1 out (no relu); layers 1..2 inputs are
    // previous hidden outputs (relu folded into A-load)
    for (int l = 0; l < L; ++l) {
        const float* b = b_h + (size_t)l * HID;
        if (l == 0)
            gemm_wmma_kernel<HID, false><<<gemmBlocks, T, 0, stream>>>(O, WpHs[l], B1, N);
        else
            gemm_wmma_kernel<HID, true ><<<gemmBlocks, T, 0, stream>>>(O, WpHs[l], B1, N);
        agg_init_kernel<<<initBlocks, T, 0, stream>>>(B1, dinv, b, O, totalNC);
        agg_edges_kernel<<<edgeBlocks, T, 0, stream>>>(B1, src, dst, dinv, O, E);
    }

    // output conv: relu'd input, no bias
    gemm_wmma_kernel<HID, true><<<gemmBlocks, T, 0, stream>>>(O, WpOut, B1, N);
    agg_init_kernel<<<initBlocks, T, 0, stream>>>(B1, dinv, nullptr, O, totalNC);
    agg_edges_kernel<<<edgeBlocks, T, 0, stream>>>(B1, src, dst, dinv, O, E);
}
